// HNHNModel_69818988364049
// MI455X (gfx1250) — matrix operand order repaired and checked by reading, not compiled, not run on hardware
//
#include <hip/hip_runtime.h>
#include <hip/hip_bf16.h>

// ---------------------------------------------------------------------------
// HNHN hypergraph network for MI455X (gfx1250, wave32).
// GEMMs use v_wmma_f32_16x16x32_f16 (f16 A/B staged in LDS, f32 accum),
// A fragments register-resident across all 8 N-tiles, W transposed in LDS so
// B fragments are contiguous ds_load_b128 pairs, unguarded stores into padded
// workspace rows (no EXEC juggling).
// SpMMs use f32 global atomics (feature matrices are L2-resident: 51MB << 192MB).
// ---------------------------------------------------------------------------

#define N_NODES 100000
#define N_EDGES 50000
#define NNZ_CNT 400000
#define IN_CH   14
#define HID     128

#define PAD_NODE_ROWS (((N_NODES + 63) / 64) * 64)   // 100032
#define PAD_EDGE_ROWS (((N_EDGES + 63) / 64) * 64)   // 50048

typedef __attribute__((ext_vector_type(16))) _Float16 v16h;
typedef __attribute__((ext_vector_type(8)))  float    v8f;

union F16Frag { v16h v; _Float16 e[16]; };

// ---------------------------------------------------------------------------
__global__ void k_zero(float* __restrict__ p, long long n) {
    long long i = (long long)blockIdx.x * blockDim.x + threadIdx.x;
    long long s = (long long)gridDim.x * blockDim.x;
    for (; i < n; i += s) p[i] = 0.0f;
}

// degrees: segment sums of inc_vals over node_idx / edge_idx
__global__ void k_deg(const float* __restrict__ inc, const int* __restrict__ nidx,
                      const int* __restrict__ eidx, float* __restrict__ node_deg,
                      float* __restrict__ edge_deg, int nnz) {
    int i = blockIdx.x * blockDim.x + threadIdx.x;
    if (i < nnz) {
        float v = inc[i];
        atomicAdd(&node_deg[nidx[i]], v);
        atomicAdd(&edge_deg[eidx[i]], v);
    }
}

// out = d > 0 ? d^p : 0
__global__ void k_pow(const float* __restrict__ d, float* __restrict__ out, int n, float p) {
    int i = blockIdx.x * blockDim.x + threadIdx.x;
    if (i < n) {
        float v = d[i];
        out[i] = (v > 0.0f) ? __powf(v, p) : 0.0f;
    }
}

// d0l[node] += inc*edge_card[edge];  d1l[edge] += inc*node_card[node]
__global__ void k_dl(const float* __restrict__ inc, const int* __restrict__ nidx,
                     const int* __restrict__ eidx, const float* __restrict__ ecard,
                     const float* __restrict__ ncard, float* __restrict__ d0l,
                     float* __restrict__ d1l, int nnz) {
    int i = blockIdx.x * blockDim.x + threadIdx.x;
    if (i < nnz) {
        int n = nidx[i], e = eidx[i];
        float v = inc[i];
        atomicAdd(&d0l[n], v * ecard[e]);
        atomicAdd(&d1l[e], v * ncard[n]);
    }
}

__global__ void k_vals(const float* __restrict__ inc, const int* __restrict__ nidx,
                       const int* __restrict__ eidx, const float* __restrict__ d0l,
                       const float* __restrict__ d1l, const float* __restrict__ ecard,
                       const float* __restrict__ ncard, float* __restrict__ vals_e2n,
                       float* __restrict__ vals_n2e, int nnz) {
    int i = blockIdx.x * blockDim.x + threadIdx.x;
    if (i < nnz) {
        int n = nidx[i], e = eidx[i];
        float v  = inc[i];
        float a  = d0l[n], b = d1l[e];
        float ia = (a > 0.0f) ? 1.0f / a : 0.0f;
        float ib = (b > 0.0f) ? 1.0f / b : 0.0f;
        vals_e2n[i] = v * ia * ecard[e];
        vals_n2e[i] = v * ib * ncard[n];
    }
}

// ---------------------------------------------------------------------------
// Z[M,128] = X[M,K] @ W[K,128]  via v_wmma_f32_16x16x32_f16.
// 128 threads = 4 waves; each wave owns a 16-row tile (64 rows per block).
// KP = K padded to a multiple of 32 (compile-time).
// Z must have >= ceil(M/64)*64 rows allocated (stores are unguarded).
#define LDS_STRIDE 136  // 128 + 8 halfs pad: breaks bank conflicts

template <int KP>
__global__ __launch_bounds__(128) void k_gemm_wmma(
        const float* __restrict__ X, const float* __restrict__ W,
        float* __restrict__ Z, int M, int K) {
    __shared__ _Float16 sWt[128 * LDS_STRIDE];  // transposed: sWt[n][k]
    __shared__ _Float16 sX [64  * LDS_STRIDE];  // sX[r][k]

    constexpr int NKT    = KP / 32;
    constexpr int KSHIFT = (KP == 32) ? 5 : 7;

    const int tid  = threadIdx.x;
    const int lane = tid & 31;
    const int wave = tid >> 5;
    const int rowBlock = blockIdx.x * 64;

    // stage W [K x 128] transposed -> sWt[n*LDS_STRIDE + k], zero-padded to KP
    for (int idx = tid; idx < KP * 128; idx += 128) {
        int k = idx >> 7, n = idx & 127;
        float v = (k < K) ? W[k * 128 + n] : 0.0f;
        sWt[n * LDS_STRIDE + k] = (_Float16)v;
    }
    // stage X tile [64 x K] -> sX[r*LDS_STRIDE + k], zero-padded
    for (int idx = tid; idx < (KP << 6); idx += 128) {
        int r = idx >> KSHIFT, k = idx & (KP - 1);
        int row = rowBlock + r;
        float v = (row < M && k < K) ? X[(size_t)row * K + k] : 0.0f;
        sX[r * LDS_STRIDE + k] = (_Float16)v;
    }
    __syncthreads();

    const int m  = lane & 15;
    const int hi = lane >> 4;  // half-wave select (0/1)
    const _Float16* xrow = &sX[(wave * 16 + m) * LDS_STRIDE];

    // A fragments: identical for all 8 N-tiles -> load once, keep in VGPRs.
    // A 16x32 f16 layout: lanes 0-15 rows M: e[0..7]=K0..7, e[8..15]=K16..23;
    //                     lanes 16-31 same rows: e[0..7]=K8..15, e[8..15]=K24..31.
    F16Frag afrag[NKT];
#pragma unroll
    for (int kk = 0; kk < NKT; ++kk) {
        const int kt = kk * 32;
#pragma unroll
        for (int j = 0; j < 8; ++j) {
            afrag[kk].e[j]     = xrow[kt + hi * 8 + j];        // 16B contiguous
            afrag[kk].e[8 + j] = xrow[kt + 16 + hi * 8 + j];   // 16B contiguous
        }
    }

    const int rowBase = rowBlock + wave * 16 + hi * 8;

#pragma unroll
    for (int nt = 0; nt < 8; ++nt) {
        const int ncol = nt * 16 + (lane & 15);
        // B 32x16 f16 layout: lane col = lane&15; lanes 0-15 hold K=kt..kt+7,kt+16..kt+23,
        // lanes 16-31 hold K=kt+8..kt+15,kt+24..kt+31 -> two 16B chunks from sWt.
        v8f acc = {};
#pragma unroll
        for (int kk = 0; kk < NKT; ++kk) {
            const _Float16* wcol = &sWt[ncol * LDS_STRIDE + kk * 32 + hi * 8];
            F16Frag b;
#pragma unroll
            for (int j = 0; j < 16; ++j) b.e[j] = wcol[j < 8 ? j : j + 8];     // 2x ds_load_b128
            acc = __builtin_amdgcn_wmma_f32_16x16x32_f16(
                      false, afrag[kk].v, false, b.v, (short)0, acc, false, false);
        }
        // C/D 16x16 f32: lane<16 -> N=lane, M=r; lane>=16 -> N=lane-16, M=r+8.
        // Unguarded: Z rows are padded to the block grid.
        float* zrow = Z + (size_t)rowBase * HID + ncol;
#pragma unroll
        for (int r = 0; r < 8; ++r) zrow[(size_t)r * HID] = acc[r];
    }
}

// ---------------------------------------------------------------------------
// Acc[sidx[i], :] += vals[i] * Xin[gidx[i], :]   (one wave per nnz entry)
__global__ void k_spmm(const float* __restrict__ vals, const int* __restrict__ gidx,
                       const int* __restrict__ sidx, const float* __restrict__ Xin,
                       float* __restrict__ Acc, int nnz) {
    int lane   = threadIdx.x & 31;
    int wave   = (blockIdx.x * blockDim.x + threadIdx.x) >> 5;
    int nwaves = (gridDim.x * blockDim.x) >> 5;
    for (int i = wave; i < nnz; i += nwaves) {
        float v = vals[i];
        int g = gidx[i], s = sidx[i];
        const float4* src = (const float4*)(Xin + (size_t)g * 128);
        float4 x = src[lane];
        float* dst = Acc + (size_t)s * 128 + lane * 4;
        atomicAdd(dst + 0, v * x.x);
        atomicAdd(dst + 1, v * x.y);
        atomicAdd(dst + 2, v * x.z);
        atomicAdd(dst + 3, v * x.w);
    }
}

__global__ void k_bias_relu(float* __restrict__ X, const float* __restrict__ b,
                            long long total) {
    long long i = (long long)blockIdx.x * blockDim.x + threadIdx.x;
    long long s = (long long)gridDim.x * blockDim.x;
    for (; i < total; i += s) {
        int c = (int)(i & 127);
        X[i] = fmaxf(X[i] + b[c], 0.0f);
    }
}

// max over rows; H >= 0 after ReLU so uint bit pattern is order-preserving
__global__ void k_maxpool(const float* __restrict__ H, unsigned int* __restrict__ bits,
                          int M) {
    int c = threadIdx.x;  // 128 threads
    float m = 0.0f;
    for (int r = blockIdx.x; r < M; r += gridDim.x)
        m = fmaxf(m, H[(size_t)r * 128 + c]);
    atomicMax(&bits[c], __float_as_uint(m));
}

__global__ void k_final(const unsigned int* __restrict__ bits,
                        const float* __restrict__ Wl, const float* __restrict__ bl,
                        float* __restrict__ out) {
    __shared__ float red[128];
    int t = threadIdx.x;
    red[t] = __uint_as_float(bits[t]) * Wl[t];
    __syncthreads();
    for (int s = 64; s > 0; s >>= 1) {
        if (t < s) red[t] += red[t + s];
        __syncthreads();
    }
    if (t == 0) out[0] = red[0] + bl[0];
}

// ---------------------------------------------------------------------------
extern "C" void kernel_launch(void* const* d_in, const int* in_sizes, int n_in,
                              void* d_out, int out_size, void* d_ws, size_t ws_size,
                              hipStream_t stream) {
    (void)in_sizes; (void)n_in; (void)out_size; (void)ws_size;

    const float* x0    = (const float*)d_in[0];
    const float* inc   = (const float*)d_in[1];
    const float* W01_0 = (const float*)d_in[2];
    const float* b1_0  = (const float*)d_in[3];
    const float* W10_0 = (const float*)d_in[4];
    const float* b0_0  = (const float*)d_in[5];
    const float* W01_1 = (const float*)d_in[6];
    const float* b1_1  = (const float*)d_in[7];
    const float* W10_1 = (const float*)d_in[8];
    const float* b0_1  = (const float*)d_in[9];
    const float* W_lin = (const float*)d_in[10];
    const float* b_lin = (const float*)d_in[11];
    const int*   nidx  = (const int*)d_in[12];
    const int*   eidx  = (const int*)d_in[13];
    float*       out   = (float*)d_out;

    // workspace carve-up (floats); bufA/bufC padded so GEMM stores need no guard
    float* ws        = (float*)d_ws;
    float* edge_deg  = ws;                                 // E
    float* node_deg  = edge_deg + N_EDGES;                 // N
    float* edge_card = node_deg + N_NODES;                 // E
    float* node_card = edge_card + N_EDGES;                // N
    float* d0l       = node_card + N_NODES;                // N
    float* d1l       = d0l + N_NODES;                      // E
    float* vals_e2n  = d1l + N_EDGES;                      // NNZ
    float* vals_n2e  = vals_e2n + NNZ_CNT;                 // NNZ
    float* bufA      = vals_n2e + NNZ_CNT;                 // PAD_NODE_ROWS*128
    float* bufB      = bufA + (size_t)PAD_NODE_ROWS * HID; // E*128
    float* bufC      = bufB + (size_t)N_EDGES * HID;       // PAD_NODE_ROWS*128
    unsigned int* pooled = (unsigned int*)(bufC + (size_t)PAD_NODE_ROWS * HID); // 128

    const int T = 256;
    const int nnzB = (NNZ_CNT + T - 1) / T;
    const long long nodeF = (long long)N_NODES * HID;
    const long long edgeF = (long long)N_EDGES * HID;
    const dim3 zeroG(2048), zeroB(T);
    const dim3 spmmG(2048), spmmB(T);
    const int gemmNodeBlocks = PAD_NODE_ROWS / 64;
    const int gemmEdgeBlocks = PAD_EDGE_ROWS / 64;

    // ---- normalization ----
    k_zero<<<zeroG, zeroB, 0, stream>>>(edge_deg, (long long)(2 * (N_EDGES + N_NODES)
                                                  + N_NODES + N_EDGES));
    k_deg<<<nnzB, T, 0, stream>>>(inc, nidx, eidx, node_deg, edge_deg, NNZ_CNT);
    k_pow<<<(N_EDGES + T - 1) / T, T, 0, stream>>>(edge_deg, edge_card, N_EDGES, -1.5f);
    k_pow<<<(N_NODES + T - 1) / T, T, 0, stream>>>(node_deg, node_card, N_NODES, -0.5f);
    k_dl<<<nnzB, T, 0, stream>>>(inc, nidx, eidx, edge_card, node_card, d0l, d1l, NNZ_CNT);
    k_vals<<<nnzB, T, 0, stream>>>(inc, nidx, eidx, d0l, d1l, edge_card, node_card,
                                   vals_e2n, vals_n2e, NNZ_CNT);

    // ---- layer 1 ----
    k_gemm_wmma<32><<<gemmNodeBlocks, 128, 0, stream>>>(x0, W01_0, bufA, N_NODES, IN_CH);
    k_zero<<<zeroG, zeroB, 0, stream>>>(bufB, edgeF);
    k_spmm<<<spmmG, spmmB, 0, stream>>>(vals_n2e, nidx, eidx, bufA, bufB, NNZ_CNT);
    k_bias_relu<<<zeroG, zeroB, 0, stream>>>(bufB, b1_0, edgeF);
    k_gemm_wmma<128><<<gemmEdgeBlocks, 128, 0, stream>>>(bufB, W10_0, bufC, N_EDGES, HID);
    k_zero<<<zeroG, zeroB, 0, stream>>>(bufA, nodeF);
    k_spmm<<<spmmG, spmmB, 0, stream>>>(vals_e2n, eidx, nidx, bufC, bufA, NNZ_CNT);
    k_bias_relu<<<zeroG, zeroB, 0, stream>>>(bufA, b0_0, nodeF);

    // ---- layer 2 ----
    k_gemm_wmma<128><<<gemmNodeBlocks, 128, 0, stream>>>(bufA, W01_1, bufC, N_NODES, HID);
    k_zero<<<zeroG, zeroB, 0, stream>>>(bufB, edgeF);
    k_spmm<<<spmmG, spmmB, 0, stream>>>(vals_n2e, nidx, eidx, bufC, bufB, NNZ_CNT);
    k_bias_relu<<<zeroG, zeroB, 0, stream>>>(bufB, b1_1, edgeF);
    k_gemm_wmma<128><<<gemmEdgeBlocks, 128, 0, stream>>>(bufB, W10_1, bufC, N_EDGES, HID);
    k_zero<<<zeroG, zeroB, 0, stream>>>(bufA, nodeF);
    k_spmm<<<spmmG, spmmB, 0, stream>>>(vals_e2n, eidx, nidx, bufC, bufA, NNZ_CNT);
    k_bias_relu<<<zeroG, zeroB, 0, stream>>>(bufA, b0_1, nodeF);

    // ---- pool + head ----
    k_zero<<<1, 128, 0, stream>>>((float*)pooled, 128);
    k_maxpool<<<512, 128, 0, stream>>>(bufA, pooled, N_NODES);
    k_final<<<1, 128, 0, stream>>>(pooled, W_lin, b_lin, out);
}
